// CausalSelfAttention_14139032339026
// MI455X (gfx1250) — matrix operand order repaired and checked by reading, not compiled, not run on hardware
//
#include <hip/hip_runtime.h>
#include <hip/hip_bf16.h>

typedef __attribute__((ext_vector_type(16))) _Float16 v16h;
typedef __attribute__((ext_vector_type(8)))  float    v8f;

// ---------------------------------------------------------------------------
// Problem constants (reference: B=4, T=2048, C=1024, H=16, D=64)
// ---------------------------------------------------------------------------
#define BB   4
#define TT   2048
#define CC   1024
#define HH   16
#define DD   64
#define M_ROWS (BB * TT)        // 8192
#define N_QKV  (3 * CC)         // 3072

__device__ __forceinline__ v8f wmma_f16(v16h a, v16h b, v8f c) {
  // D = A(16x32 f16) * B(32x16 f16) + C(16x16 f32)
  return __builtin_amdgcn_wmma_f32_16x16x32_f16(
      /*neg_a=*/false, a, /*neg_b=*/false, b,
      /*c_mod=*/(short)0, c, /*reuse_a=*/false, /*reuse_b=*/false);
}

// Load an A-matrix fragment (16x32 f16) for WMMA from a row-major f16 matrix.
// Layout: lane L (L&15 = M row); element pair j holds
//   K = (lane>>4)*16 + 2*j + {0,1}
// The 8 pairs are 32-byte contiguous (matches the contiguous B loads).
__device__ __forceinline__ v16h load_a_frag(const _Float16* base, int ld,
                                            int row, int k0, int hl) {
  const _Float16* p = base + (size_t)row * ld + k0 + hl * 16;
  v16h a;
#pragma unroll
  for (int j = 0; j < 8; ++j) {
    int kb = 2 * j;
    union { unsigned u; _Float16 h[2]; } t;
    t.u = *(const unsigned*)(p + kb);
    a[2 * j]     = t.h[0];
    a[2 * j + 1] = t.h[1];
  }
  return a;
}

// ---------------------------------------------------------------------------
// Precision conversion helpers
// ---------------------------------------------------------------------------
__global__ void cvt_f32_to_f16_kernel(const float* __restrict__ in,
                                      _Float16* __restrict__ out, int n) {
  int i = blockIdx.x * 256 + threadIdx.x;
  if (i < n) out[i] = (_Float16)in[i];
}

// in: [K, N] f32 row-major  ->  out: [N, K] f16 row-major (transpose)
__global__ void transpose_cvt_kernel(const float* __restrict__ in,
                                     _Float16* __restrict__ out, int K, int N) {
  int idx = blockIdx.x * 256 + threadIdx.x;   // idx = n*K + k (coalesced writes)
  int k = idx % K;
  int n = idx / K;
  if (n < N) out[idx] = (_Float16)in[(size_t)k * N + n];
}

// ---------------------------------------------------------------------------
// Fragment bundles for the 32x64-tile GEMMs (2 A frags, 4 B frags per k-step)
// ---------------------------------------------------------------------------
struct FragA2 { v16h a[2]; };
struct FragB4 { v16h b[4]; };

__device__ __forceinline__ FragA2 load_a2(const _Float16* A, int ld, int m0,
                                          int k0, int col, int hl) {
  FragA2 f;
  f.a[0] = load_a_frag(A, ld, m0 + col, k0, hl);
  f.a[1] = load_a_frag(A, ld, m0 + 16 + col, k0, hl);
  return f;
}

__device__ __forceinline__ FragB4 load_b4(const _Float16* Bt, int ld, int n0,
                                          int k0, int col, int hl) {
  FragB4 f;
#pragma unroll
  for (int g = 0; g < 4; ++g)
    f.b[g] = *(const v16h*)(Bt + (size_t)(n0 + 16 * g + col) * ld + k0 + hl * 16);
  return f;
}

__device__ __forceinline__ void mul_acc(v8f (&acc)[2][4], const FragA2& fa,
                                        const FragB4& fb) {
#pragma unroll
  for (int mi = 0; mi < 2; ++mi)
#pragma unroll
    for (int g = 0; g < 4; ++g)
      acc[mi][g] = wmma_f16(fa.a[mi], fb.b[g], acc[mi][g]);
}

// ---------------------------------------------------------------------------
// QKV GEMM: [8192,1024] x [1024,3072] -> scatter into Q/K (B,H,T,D) and
// V^T (B,H,D,T), all f16.  32x64 tile per wave; two fragment sets pipelined
// in-place (no rotation copies); waves-per-eu=1 so nothing spills.
// ---------------------------------------------------------------------------
__global__ __launch_bounds__(256, 1)
void qkv_gemm_kernel(const _Float16* __restrict__ xh,
                     const _Float16* __restrict__ WaT,  // [3072,1024]
                     const float* __restrict__ bias,
                     _Float16* __restrict__ qbuf,
                     _Float16* __restrict__ kbuf,
                     _Float16* __restrict__ vbufT) {
  int lane = threadIdx.x & 31;
  int wrp  = threadIdx.x >> 5;
  int hl   = lane >> 4;
  int col  = lane & 15;
  int wid  = blockIdx.x * 8 + wrp;            // 0 .. 12287
  int ntile = wid % (N_QKV / 64);             // 48 column tiles of 64
  int mtile = wid / (N_QKV / 64);             // 256 row tiles of 32
  int m0 = mtile * 32;
  int n0 = ntile * 64;

  v8f acc[2][4] = {};

  FragA2 a0 = load_a2(xh, CC, m0, 0, col, hl);
  FragB4 b0 = load_b4(WaT, CC, n0, 0, col, hl);
  FragA2 a1 = load_a2(xh, CC, m0, 32, col, hl);
  FragB4 b1 = load_b4(WaT, CC, n0, 32, col, hl);
  for (int k0 = 0; k0 < CC; k0 += 64) {
    mul_acc(acc, a0, b0);                     // consumes set 0
    if (k0 + 64 < CC) {                       // refill set 0 under set-1 wmma
      a0 = load_a2(xh, CC, m0, k0 + 64, col, hl);
      b0 = load_b4(WaT, CC, n0, k0 + 64, col, hl);
    }
    mul_acc(acc, a1, b1);                     // consumes set 1
    if (k0 + 96 < CC) {                       // refill set 1 under set-0 wmma
      a1 = load_a2(xh, CC, m0, k0 + 96, col, hl);
      b1 = load_b4(WaT, CC, n0, k0 + 96, col, hl);
    }
  }

  int region = n0 >> 10;                      // 0 = Q, 1 = K, 2 = V
#pragma unroll
  for (int g = 0; g < 4; ++g) {
    int n = n0 + 16 * g + col;
    int c = n & (CC - 1);
    int hh = c >> 6;                          // head
    int d  = c & (DD - 1);                    // dim in head
    float bv = bias[n];
#pragma unroll
    for (int mi = 0; mi < 2; ++mi) {
#pragma unroll
      for (int v = 0; v < 8; ++v) {
        int m = m0 + 16 * mi + v + 8 * hl;    // global token row
        int bidx = m >> 11;                   // / TT
        int t    = m & (TT - 1);
        float val = acc[mi][g][v] + bv;
        size_t bh = (size_t)(bidx * HH + hh);
        if (region == 0)
          qbuf[(bh * TT + t) * DD + d] = (_Float16)val;
        else if (region == 1)
          kbuf[(bh * TT + t) * DD + d] = (_Float16)val;
        else
          vbufT[(bh * DD + d) * TT + t] = (_Float16)val;
      }
    }
  }
}

// ---------------------------------------------------------------------------
// Flash attention: one wave per (b, h, 16-row q tile); 32 keys per iteration.
// V frags + next K frags are issued before the softmax VALU section so memory
// latency hides under exp/shuffle work.  P staged through LDS for the
// C-fragment -> A-fragment relayout.
// ---------------------------------------------------------------------------
#define PSTRIDE 36    // halves per staged row (16x32 tile, padded)

__global__ __launch_bounds__(256, 1)
void attn_kernel(const _Float16* __restrict__ qbuf,
                 const _Float16* __restrict__ kbuf,
                 const _Float16* __restrict__ vbufT,
                 _Float16* __restrict__ ybuf) {
  __shared__ _Float16 smem[8 * 16 * PSTRIDE];
  int lane = threadIdx.x & 31;
  int wrp  = threadIdx.x >> 5;
  int hl   = lane >> 4;
  int col  = lane & 15;
  int wid  = blockIdx.x * 8 + wrp;           // 0 .. 8191
  int qtile = wid & 127;
  int h     = (wid >> 7) & (HH - 1);
  int b     = wid >> 11;

  size_t bh = (size_t)(b * HH + h);
  const _Float16* Q  = qbuf  + bh * TT * DD;
  const _Float16* K  = kbuf  + bh * TT * DD;
  const _Float16* Vt = vbufT + bh * DD * TT;
  _Float16* pstage = smem + wrp * (16 * PSTRIDE);

  int q0 = qtile * 16;
  v16h aq0 = load_a_frag(Q, DD, q0 + col, 0, hl);
  v16h aq1 = load_a_frag(Q, DD, q0 + col, 32, hl);

  v8f acc[4] = {};
  float mrow[8], lrow[8];
#pragma unroll
  for (int v = 0; v < 8; ++v) { mrow[v] = -1e30f; lrow[v] = 0.0f; }

  const float scale = 0.125f;                // 1/sqrt(64)
  int nkv = q0 + 16;                         // keys 0 .. q0+15 valid

  // K fragments for the current 32-key block: [sub0 k0, sub0 k32, sub1 k0, sub1 k32]
  v16h kf[4];
  {
    const _Float16* kp0 = K + (size_t)(col) * DD + hl * 16;
    const _Float16* kp1 = K + (size_t)(16 + col) * DD + hl * 16;
    kf[0] = *(const v16h*)kp0;
    kf[1] = *(const v16h*)(kp0 + 32);
    kf[2] = *(const v16h*)kp1;
    kf[3] = *(const v16h*)(kp1 + 32);
  }

  for (int j = 0; j < nkv; j += 32) {
    // --- S = Q * K^T for 32 keys (two 16-key subtiles), K-dim 64 ---
    v8f s0 = {}, s1 = {};
    s0 = wmma_f16(aq0, kf[0], s0);
    s0 = wmma_f16(aq1, kf[1], s0);
    s1 = wmma_f16(aq0, kf[2], s1);
    s1 = wmma_f16(aq1, kf[3], s1);

    // --- issue V frags for this block + K frags for next block early ---
    v16h vf[4];
#pragma unroll
    for (int g = 0; g < 4; ++g)
      vf[g] = *(const v16h*)(Vt + (size_t)(16 * g + col) * TT + j + hl * 16);
    if (j + 32 < nkv) {
      const _Float16* kp0 = K + (size_t)(j + 32 + col) * DD + hl * 16;
      const _Float16* kp1 = K + (size_t)(j + 48 + col) * DD + hl * 16;
      kf[0] = *(const v16h*)kp0;
      kf[1] = *(const v16h*)(kp0 + 32);
      kf[2] = *(const v16h*)kp1;
      kf[3] = *(const v16h*)(kp1 + 32);
    }

    // --- causal mask + online softmax (rows live in 16-lane halves) ---
#pragma unroll
    for (int v = 0; v < 8; ++v) {
      int r  = v + 8 * hl;                   // row within tile
      int qg = q0 + r;                       // global query index
      float x0 = s0[v] * scale; if (j + col > qg)      x0 = -1e30f;
      float x1 = s1[v] * scale; if (j + 16 + col > qg) x1 = -1e30f;
      float mx = fmaxf(x0, x1);
      mx = fmaxf(mx, __shfl_xor(mx, 1, 32));
      mx = fmaxf(mx, __shfl_xor(mx, 2, 32));
      mx = fmaxf(mx, __shfl_xor(mx, 4, 32));
      mx = fmaxf(mx, __shfl_xor(mx, 8, 32));
      float mnew = fmaxf(mrow[v], mx);
      float p0 = __expf(x0 - mnew);
      float p1 = __expf(x1 - mnew);
      float corr = __expf(mrow[v] - mnew);
      mrow[v] = mnew;
      float rs = p0 + p1;
      rs += __shfl_xor(rs, 1, 32);
      rs += __shfl_xor(rs, 2, 32);
      rs += __shfl_xor(rs, 4, 32);
      rs += __shfl_xor(rs, 8, 32);
      lrow[v] = lrow[v] * corr + rs;
      acc[0][v] *= corr; acc[1][v] *= corr;
      acc[2][v] *= corr; acc[3][v] *= corr;
      pstage[r * PSTRIDE + col]      = (_Float16)p0;
      pstage[r * PSTRIDE + 16 + col] = (_Float16)p1;
    }
    asm volatile("s_wait_dscnt 0" ::: "memory");

    // --- acc += P(16x32) * V(32x64) ---
    v16h ap = load_a_frag(pstage, PSTRIDE, col, 0, hl);
#pragma unroll
    for (int g = 0; g < 4; ++g)
      acc[g] = wmma_f16(ap, vf[g], acc[g]);
  }

  // --- normalize and store y in [B, T, C] f16 ---
#pragma unroll
  for (int v = 0; v < 8; ++v) {
    float inv = 1.0f / lrow[v];
    int qg = q0 + v + 8 * hl;
    size_t rowoff = ((size_t)b * TT + qg) * CC + h * DD;
#pragma unroll
    for (int g = 0; g < 4; ++g)
      ybuf[rowoff + 16 * g + col] = (_Float16)(acc[g][v] * inv);
  }
}

// ---------------------------------------------------------------------------
// Output projection: [8192,1024] x [1024,1024] + bias -> fp32 out
// 32x64 tile per wave, pipelined identically to the QKV GEMM.
// ---------------------------------------------------------------------------
__global__ __launch_bounds__(256, 1)
void proj_gemm_kernel(const _Float16* __restrict__ yh,
                      const _Float16* __restrict__ WpT, // [1024,1024] NxK
                      const float* __restrict__ bias,
                      float* __restrict__ out) {
  int lane = threadIdx.x & 31;
  int wrp  = threadIdx.x >> 5;
  int hl   = lane >> 4;
  int col  = lane & 15;
  int wid  = blockIdx.x * 8 + wrp;            // 0 .. 4095
  int ntile = wid % (CC / 64);                // 16 column tiles of 64
  int mtile = wid / (CC / 64);                // 256 row tiles of 32
  int m0 = mtile * 32;
  int n0 = ntile * 64;

  v8f acc[2][4] = {};

  FragA2 a0 = load_a2(yh, CC, m0, 0, col, hl);
  FragB4 b0 = load_b4(WpT, CC, n0, 0, col, hl);
  FragA2 a1 = load_a2(yh, CC, m0, 32, col, hl);
  FragB4 b1 = load_b4(WpT, CC, n0, 32, col, hl);
  for (int k0 = 0; k0 < CC; k0 += 64) {
    mul_acc(acc, a0, b0);
    if (k0 + 64 < CC) {
      a0 = load_a2(yh, CC, m0, k0 + 64, col, hl);
      b0 = load_b4(WpT, CC, n0, k0 + 64, col, hl);
    }
    mul_acc(acc, a1, b1);
    if (k0 + 96 < CC) {
      a1 = load_a2(yh, CC, m0, k0 + 96, col, hl);
      b1 = load_b4(WpT, CC, n0, k0 + 96, col, hl);
    }
  }

#pragma unroll
  for (int g = 0; g < 4; ++g) {
    int n = n0 + 16 * g + col;
    float bv = bias[n];
#pragma unroll
    for (int mi = 0; mi < 2; ++mi)
#pragma unroll
      for (int v = 0; v < 8; ++v) {
        int m = m0 + 16 * mi + v + 8 * hl;
        out[(size_t)m * CC + n] = acc[mi][g][v] + bv;
      }
  }
}

// ---------------------------------------------------------------------------
// Host-side launch
// ---------------------------------------------------------------------------
extern "C" void kernel_launch(void* const* d_in, const int* in_sizes, int n_in,
                              void* d_out, int out_size, void* d_ws, size_t ws_size,
                              hipStream_t stream) {
  const float* x      = (const float*)d_in[0];   // [B,T,C]
  const float* W_attn = (const float*)d_in[1];   // [C,3C]
  const float* b_attn = (const float*)d_in[2];   // [3C]
  const float* W_proj = (const float*)d_in[3];   // [C,C]
  const float* b_proj = (const float*)d_in[4];   // [C]
  float* out = (float*)d_out;

  char* ws = (char*)d_ws;
  size_t off = 0;
  auto carve = [&](size_t bytes) -> void* {
    void* p = ws + off;
    off += (bytes + 255) & ~(size_t)255;
    return p;
  };
  _Float16* xh    = (_Float16*)carve((size_t)M_ROWS * CC * 2);   // x in f16
  _Float16* WaT   = (_Float16*)carve((size_t)N_QKV * CC * 2);    // W_attn^T f16
  _Float16* WpT   = (_Float16*)carve((size_t)CC * CC * 2);       // W_proj^T f16
  _Float16* qbuf  = (_Float16*)carve((size_t)BB * HH * TT * DD * 2);
  _Float16* kbuf  = (_Float16*)carve((size_t)BB * HH * TT * DD * 2);
  _Float16* vbufT = (_Float16*)carve((size_t)BB * HH * DD * TT * 2);
  _Float16* ybuf  = (_Float16*)carve((size_t)M_ROWS * CC * 2);

  // 1) conversions
  {
    int n = M_ROWS * CC;  // 8388608
    cvt_f32_to_f16_kernel<<<n / 256, 256, 0, stream>>>(x, xh, n);
  }
  {
    int total = CC * N_QKV;  // 3145728
    transpose_cvt_kernel<<<total / 256, 256, 0, stream>>>(W_attn, WaT, CC, N_QKV);
  }
  {
    int total = CC * CC;     // 1048576
    transpose_cvt_kernel<<<total / 256, 256, 0, stream>>>(W_proj, WpT, CC, CC);
  }

  // 2) QKV projection: 256 * 48 = 12288 waves, 8 waves/block
  qkv_gemm_kernel<<<(M_ROWS / 32) * (N_QKV / 64) / 8, 256, 0, stream>>>(
      xh, WaT, b_attn, qbuf, kbuf, vbufT);

  // 3) flash attention: B*H*(T/16) = 8192 waves, 8 waves/block
  attn_kernel<<<BB * HH * (TT / 16) / 8, 256, 0, stream>>>(qbuf, kbuf, vbufT, ybuf);

  // 4) output projection: 256 * 16 = 4096 waves, 8 waves/block
  proj_gemm_kernel<<<(M_ROWS / 32) * (CC / 64) / 8, 256, 0, stream>>>(
      ybuf, WpT, b_proj, out);
}
